// MoeFusedLinear_4088808865950
// MI455X (gfx1250) — compile-verified, matrix operand order checked by reading, not run on hardware
//
#include <hip/hip_runtime.h>
#include <hip/hip_bf16.h>

// ---- ext-vector types (trivial, union-safe) --------------------------------
typedef __attribute__((ext_vector_type(16))) __bf16 v16bf;
typedef __attribute__((ext_vector_type(8)))  float  v8f;
typedef __attribute__((ext_vector_type(4)))  float  f32x4;
typedef __attribute__((ext_vector_type(4)))  __bf16 bf16x4;

#define NUM_E   8
#define IN_F    2048
#define OUT_F   2048
#define TOT     2048
#define BM      128
#define BN      128
#define BK      64          // bf16 K elements per stage
#define PITCH   72          // bf16 elems per LDS row (144B) -> bank-conflict pad
#define KSTEPS  (IN_F / BK) // 32

union FragU { v16bf v; f32x4 q[2]; };

__global__ __launch_bounds__(256)
void moe_grouped_gemm_bf16_wmma(const float* __restrict__ x,
                                const float* __restrict__ w,
                                const int*   __restrict__ m_sizes,
                                float*       __restrict__ out) {
  // double-buffered bf16 tiles: 2 * 2 * 128*72*2B = 72 KB (WGP has 320 KB)
  __shared__ __align__(16) __bf16 lsA[2][BM * PITCH];
  __shared__ __align__(16) __bf16 lsB[2][BM * PITCH];

  const int e     = blockIdx.z;
  const int tileM = blockIdx.y;       // M tile within this expert's token block
  const int n0    = blockIdx.x * BN;  // output-feature tile base

  int start = 0;
  for (int i = 0; i < e; ++i) start += m_sizes[i];
  const int msize = m_sizes[e];

  const int tid   = threadIdx.x;
  const int lane  = tid & 31;
  const int wid   = tid >> 5;
  const int waveN = wid & 1;     // 2 waves across N, 64 cols each
  const int waveM = wid >> 1;    // 4 waves across M, 32 rows each
  const int half  = (lane >> 4) & 1;
  const int lr    = lane & 15;

  const float* wexp = w + (size_t)e * OUT_F * IN_F;

  v8f acc[2][4];
  {
    v8f zero = {0.f, 0.f, 0.f, 0.f, 0.f, 0.f, 0.f, 0.f};
#pragma unroll
    for (int mi = 0; mi < 2; ++mi)
#pragma unroll
      for (int ni = 0; ni < 4; ++ni) acc[mi][ni] = zero;
  }

  // staging: 128x64 fp32 tile = 2048 f32x4 -> 8 f32x4 per thread per matrix
  f32x4 pa[8], pb[8];

  auto g_load = [&](int k0) {
#pragma unroll
    for (int i = 0; i < 8; ++i) {
      const int idx = tid + i * 256;
      const int row = idx >> 4;      // 0..127
      const int c4  = idx & 15;      // 0..15 (x4 floats)
      int arow = start + tileM * BM + row;
      arow = arow < TOT ? arow : (TOT - 1);
      pa[i] = *(const f32x4*)(x    + (size_t)arow * IN_F + k0 + c4 * 4);
      const int brow = n0 + row;
      pb[i] = *(const f32x4*)(wexp + (size_t)brow * IN_F + k0 + c4 * 4);
    }
  };

  auto l_store = [&](int buf) {
#pragma unroll
    for (int i = 0; i < 8; ++i) {
      const int idx = tid + i * 256;
      const int row = idx >> 4;
      const int c4  = idx & 15;
      bf16x4 a4 = { (__bf16)pa[i].x, (__bf16)pa[i].y, (__bf16)pa[i].z, (__bf16)pa[i].w };
      bf16x4 b4 = { (__bf16)pb[i].x, (__bf16)pb[i].y, (__bf16)pb[i].z, (__bf16)pb[i].w };
      *(bf16x4*)(&lsA[buf][row * PITCH + c4 * 4]) = a4;
      *(bf16x4*)(&lsB[buf][row * PITCH + c4 * 4]) = b4;
    }
  };

  auto compute = [&](int buf) {
#pragma unroll
    for (int kc = 0; kc < BK; kc += 32) {
      // ISA 16-bit operand layout: lanes 0-15 hold K={0..7,16..23},
      // lanes 16-31 hold K={8..15,24..31} of the same row.
      FragU af[2], bfg[4];
#pragma unroll
      for (int mi = 0; mi < 2; ++mi) {
        const int row = waveM * 32 + mi * 16 + lr;
        const __bf16* p = &lsA[buf][row * PITCH + kc + half * 8];
        af[mi].q[0] = *(const f32x4*)p;          // K lo 8
        af[mi].q[1] = *(const f32x4*)(p + 16);   // K hi 8
      }
#pragma unroll
      for (int ni = 0; ni < 4; ++ni) {
        const int row = waveN * 64 + ni * 16 + lr;   // weight row n -> B column n
        const __bf16* p = &lsB[buf][row * PITCH + kc + half * 8];
        bfg[ni].q[0] = *(const f32x4*)p;
        bfg[ni].q[1] = *(const f32x4*)(p + 16);
      }
#pragma unroll
      for (int mi = 0; mi < 2; ++mi)
#pragma unroll
        for (int ni = 0; ni < 4; ++ni)
          acc[mi][ni] = __builtin_amdgcn_wmma_f32_16x16x32_bf16(
              /*neg_a=*/false, af[mi].v, /*neg_b=*/false, bfg[ni].v,
              /*c_mod=*/(short)0, acc[mi][ni],
              /*reuse_a=*/false, /*reuse_b=*/false);
    }
  };

  // ---- software pipeline: global prefetch overlaps WMMA --------------------
  g_load(0);
  l_store(0);
  __syncthreads();

#pragma unroll 2
  for (int ks = 0; ks < KSTEPS; ++ks) {
    const int cur = ks & 1;
    if (ks + 1 < KSTEPS) g_load((ks + 1) * BK);   // loads in flight...
    compute(cur);                                  // ...hidden behind 16 WMMAs
    if (ks + 1 < KSTEPS) l_store(cur ^ 1);        // cvt + ds_store to next buf
    __syncthreads();
  }

  // ---- epilogue: D layout VGPR r -> M = half*8 + r, N = lane&15 ------------
#pragma unroll
  for (int mi = 0; mi < 2; ++mi) {
#pragma unroll
    for (int ni = 0; ni < 4; ++ni) {
#pragma unroll
      for (int r = 0; r < 8; ++r) {
        const int row_local = tileM * BM + waveM * 32 + mi * 16 + half * 8 + r;
        if (row_local < msize) {
          const int col = n0 + waveN * 64 + ni * 16 + lr;
          // match reference: result materialized in bf16, returned as fp32
          const float vv = (float)(__bf16)acc[mi][ni][r];
          out[(size_t)(start + row_local) * OUT_F + col] = vv;
        }
      }
    }
  }
}

extern "C" void kernel_launch(void* const* d_in, const int* in_sizes, int n_in,
                              void* d_out, int out_size, void* d_ws, size_t ws_size,
                              hipStream_t stream) {
  const float* x  = (const float*)d_in[0];   // [2048, 2048] fp32
  const float* w  = (const float*)d_in[1];   // [8, 2048, 2048] fp32
  const int*   ms = (const int*)d_in[2];     // [8] int32
  float*       o  = (float*)d_out;           // [2048, 2048] fp32

  dim3 grid(OUT_F / BN, (TOT / NUM_E) / BM, NUM_E);  // 16 x 2 x 8 = 256 WGs
  dim3 block(256);
  hipLaunchKernelGGL(moe_grouped_gemm_bf16_wmma, grid, block, 0, stream,
                     x, w, ms, o);
}
